// CombineMultiOutputModelWeightedConcat_50577534878573
// MI455X (gfx1250) — compile-verified, matrix operand order checked by reading, not run on hardware
//
#include <hip/hip_runtime.h>
#include <cstdint>
#include <cstddef>

#define EPS 0.2f
#define BDIM 32768

typedef __bf16 bf16x16 __attribute__((ext_vector_type(16)));
typedef float f32x8 __attribute__((ext_vector_type(8)));

union FragAB { bf16x16 v; uint4 q[2]; };

__device__ inline unsigned short f2bf(float f) {
  unsigned u = __float_as_uint(f);
  unsigned r = u + 0x7FFFu + ((u >> 16) & 1u);   // round-to-nearest-even
  return (unsigned short)(r >> 16);
}
__device__ inline float bf2f(unsigned short h) {
  return __uint_as_float(((unsigned)h) << 16);
}

// ---------------------------------------------------------------------------
// WMMA GEMM: Out[B][N] = relu(A[B][K] @ W[K][N] + bias), A bf16 (stride K),
// Wt is W transposed -> [N][K] bf16. Also accumulates per-column sum / sumsq
// of the relu output into gsum/gsum2 (pre-zeroed) for BatchNorm batch stats.
// Block = 256 threads (8 waves), block tile 64x64, K stepped by 32.
// ---------------------------------------------------------------------------
__global__ __launch_bounds__(256) void gemm_bias_relu_stats(
    const unsigned short* __restrict__ A,
    const unsigned short* __restrict__ Wt,
    const float* __restrict__ bias,
    float* __restrict__ Out,
    float* __restrict__ gsum, float* __restrict__ gsum2,
    int K, int N)
{
  __shared__ unsigned short lA[64 * 40];   // padded stride 40 (80B) vs 64B rows
  __shared__ unsigned short lB[64 * 40];
  __shared__ float lsum[64];
  __shared__ float lsum2[64];

  const int t    = threadIdx.x;
  const int lane = t & 31;
  const int wave = t >> 5;
  const int mBase = blockIdx.x * 64;
  const int nBase = blockIdx.y * 64;

  if (t < 64) { lsum[t] = 0.f; lsum2[t] = 0.f; }

  const int mTile = wave & 3;   // 4 M-subtiles
  const int nPair = wave >> 2;  // 2 groups of 2 N-subtiles
  const int lhalf = lane >> 4;  // lane half (K phase per ISA 16-bit layout)
  const int l15   = lane & 15;

  f32x8 accs[2];
  #pragma unroll
  for (int i = 0; i < 2; ++i)
    accs[i] = f32x8{0.f, 0.f, 0.f, 0.f, 0.f, 0.f, 0.f, 0.f};

  const int ldRow = t >> 2;        // 0..63 (A row / Wt row(=out col))
  const int ldK   = (t & 3) * 8;   // 0,8,16,24 (bf16 elems, 16B chunks)

  for (int k0 = 0; k0 < K; k0 += 32) {
    __syncthreads();
    *(uint4*)(&lA[ldRow * 40 + ldK]) =
        *(const uint4*)(A + (size_t)(mBase + ldRow) * K + k0 + ldK);
    *(uint4*)(&lB[ldRow * 40 + ldK]) =
        *(const uint4*)(Wt + (size_t)(nBase + ldRow) * K + k0 + ldK);
    __syncthreads();

    // A fragment: row = mTile*16 + l15; lane-half selects K 0..7/8..15 then 16..23/24..31
    FragAB fa, fb0, fb1;
    {
      const unsigned short* p = &lA[(mTile * 16 + l15) * 40 + lhalf * 8];
      fa.q[0] = *(const uint4*)(p);
      fa.q[1] = *(const uint4*)(p + 16);
    }
    {
      const unsigned short* p = &lB[(nPair * 32 + l15) * 40 + lhalf * 8];
      fb0.q[0] = *(const uint4*)(p);
      fb0.q[1] = *(const uint4*)(p + 16);
      const unsigned short* p1 = p + 16 * 40;
      fb1.q[0] = *(const uint4*)(p1);
      fb1.q[1] = *(const uint4*)(p1 + 16);
    }
    accs[0] = __builtin_amdgcn_wmma_f32_16x16x32_bf16(
        false, fa.v, false, fb0.v, (short)0, accs[0], false, false);
    accs[1] = __builtin_amdgcn_wmma_f32_16x16x32_bf16(
        false, fa.v, false, fb1.v, (short)0, accs[1], false, false);
  }

  // C layout: lane col = l15, VGPR r -> row r (+8 for high lane half)
  const int rowB = mBase + mTile * 16 + lhalf * 8;
  #pragma unroll
  for (int nt = 0; nt < 2; ++nt) {
    const int colL = nPair * 32 + nt * 16 + l15;
    const int col  = nBase + colL;
    const float bv = bias[col];
    float s = 0.f, s2 = 0.f;
    #pragma unroll
    for (int r = 0; r < 8; ++r) {
      float v = accs[nt][r] + bv;
      v = v > 0.f ? v : 0.f;
      Out[(size_t)(rowB + r) * N + col] = v;
      s  += v;
      s2 += v * v;
    }
    atomicAdd(&lsum[colL], s);     // ds_add_f32
    atomicAdd(&lsum2[colL], s2);
  }
  __syncthreads();
  if (t < 64) {
    unsafeAtomicAdd(&gsum[nBase + t],  lsum[t]);    // global_atomic_add_f32
    unsafeAtomicAdd(&gsum2[nBase + t], lsum2[t]);
  }
}

// scale[c] = g[c] * rsqrt(var+EPS); shift[c] = be[c] - mean*scale[c]
__global__ void bn_finalize(const float* __restrict__ sum,
                            const float* __restrict__ sum2,
                            const float* __restrict__ g,
                            const float* __restrict__ be,
                            float* __restrict__ scale, float* __restrict__ shift,
                            int N, float invB)
{
  int i = blockIdx.x * blockDim.x + threadIdx.x;
  if (i >= N) return;
  float m = sum[i] * invB;
  float v = sum2[i] * invB - m * m;
  float s = g[i] * rsqrtf(v + EPS);
  scale[i] = s;
  shift[i] = be[i] - m * s;
}

// dst[row*dstStride + dstOff + col] = bf16(src*scale+shift)
__global__ void bn_apply_bf16(const float* __restrict__ src,
                              const float* __restrict__ scale,
                              const float* __restrict__ shift,
                              unsigned short* __restrict__ dst,
                              int N, int dstStride, int dstOff, int total)
{
  int i = blockIdx.x * blockDim.x + threadIdx.x;
  if (i >= total) return;
  int row = i / N, col = i - row * N;
  dst[(size_t)row * dstStride + dstOff + col] = f2bf(src[i] * scale[col] + shift[col]);
}

// fused BN-affine + small-N GEMV + softmax (N <= 8): one thread per row
__global__ void head_softmax(const float* __restrict__ src,   // pre-BN [B][K]
                             const float* __restrict__ scale,
                             const float* __restrict__ shift,
                             const float* __restrict__ W,     // [K][N] f32
                             const float* __restrict__ bias,  // [N]
                             float* __restrict__ out,         // [B][N]
                             int K, int N)
{
  int r = blockIdx.x * blockDim.x + threadIdx.x;
  if (r >= BDIM) return;
  float acc[8];
  for (int n = 0; n < N; ++n) acc[n] = bias[n];
  const float* x = src + (size_t)r * K;
  for (int i = 0; i < K; ++i) {
    float v = x[i] * scale[i] + shift[i];
    for (int n = 0; n < N; ++n) acc[n] += v * W[i * N + n];
  }
  float m = acc[0];
  for (int n = 1; n < N; ++n) m = fmaxf(m, acc[n]);
  float s = 0.f;
  for (int n = 0; n < N; ++n) { acc[n] = __expf(acc[n] - m); s += acc[n]; }
  float inv = 1.f / s;
  float* o = out + (size_t)r * N;
  for (int n = 0; n < N; ++n) o[n] = acc[n] * inv;
}

// xcat[b][c] *= y[b][c<256 ? 0 : 1]   (in place, bf16)
__global__ void xw_mul(unsigned short* __restrict__ xcat,
                       const float* __restrict__ y, int total)
{
  int i = blockIdx.x * blockDim.x + threadIdx.x;
  if (i >= total) return;
  int row = i >> 9, c = i & 511;
  float w = y[row * 2 + (c >> 8)];
  xcat[i] = f2bf(bf2f(xcat[i]) * w);
}

__global__ void cvt_f32_bf16(const float* __restrict__ src,
                             unsigned short* __restrict__ dst, int total)
{
  int i = blockIdx.x * blockDim.x + threadIdx.x;
  if (i >= total) return;
  dst[i] = f2bf(src[i]);
}

// W [K][N] f32 -> Wt [N][K] bf16
__global__ void wtrans(const float* __restrict__ src,
                       unsigned short* __restrict__ dst, int K, int N)
{
  int i = blockIdx.x * blockDim.x + threadIdx.x;
  if (i >= K * N) return;
  int k = i / N, n = i - k * N;
  dst[(size_t)n * K + k] = f2bf(src[i]);
}

// ---------------------------------------------------------------------------
extern "C" void kernel_launch(void* const* d_in, const int* in_sizes, int n_in,
                              void* d_out, int out_size, void* d_ws, size_t ws_size,
                              hipStream_t stream)
{
  (void)in_sizes; (void)n_in; (void)out_size; (void)ws_size;

  const float* x1f  = (const float*)d_in[0];
  const float* x2f  = (const float*)d_in[1];
  const float* W1   = (const float*)d_in[2];
  const float* b1   = (const float*)d_in[3];
  const float* W2   = (const float*)d_in[4];
  const float* b2   = (const float*)d_in[5];
  const float* g1   = (const float*)d_in[6];
  const float* be1  = (const float*)d_in[7];
  const float* g2   = (const float*)d_in[8];
  const float* be2  = (const float*)d_in[9];
  const float* g3   = (const float*)d_in[10];
  const float* be3  = (const float*)d_in[11];
  const float* g4   = (const float*)d_in[12];
  const float* be4  = (const float*)d_in[13];
  const float* Wbx1 = (const float*)d_in[14];
  const float* bbx1 = (const float*)d_in[15];
  const float* Wbx2 = (const float*)d_in[16];
  const float* bbx2 = (const float*)d_in[17];
  const float* Wwa  = (const float*)d_in[18];
  const float* bwa  = (const float*)d_in[19];
  const float* Wwb  = (const float*)d_in[20];
  const float* bwb  = (const float*)d_in[21];
  const float* Wwc  = (const float*)d_in[22];
  const float* bwc  = (const float*)d_in[23];
  const float* Wy   = (const float*)d_in[24];
  const float* by   = (const float*)d_in[25];
  const float* Wc   = (const float*)d_in[26];
  const float* bc   = (const float*)d_in[27];
  const float* WoA[4] = { (const float*)d_in[28], (const float*)d_in[30],
                          (const float*)d_in[32], (const float*)d_in[34] };
  const float* boA[4] = { (const float*)d_in[29], (const float*)d_in[31],
                          (const float*)d_in[33], (const float*)d_in[35] };
  const int NoA[4] = { 3, 3, 6, 4 };
  const size_t ooff[4] = { 0, (size_t)BDIM * 3, (size_t)BDIM * 6, (size_t)BDIM * 12 };

  // ---- workspace layout ----
  char* wsp = (char*)d_ws;
  auto alloc = [&](size_t bytes) -> char* {
    char* p = wsp; wsp += (bytes + 255) & ~(size_t)255; return p;
  };
  unsigned short* W1t   = (unsigned short*)alloc((size_t)512 * 256 * 2);
  unsigned short* W2t   = (unsigned short*)alloc((size_t)256 * 256 * 2);
  unsigned short* Wbx1t = (unsigned short*)alloc((size_t)4 * 256 * 256 * 2);
  unsigned short* Wbx2t = (unsigned short*)alloc((size_t)4 * 256 * 256 * 2);
  unsigned short* Wwat  = (unsigned short*)alloc((size_t)4 * 512 * 128 * 2);
  unsigned short* Wwbt  = (unsigned short*)alloc((size_t)4 * 128 * 64 * 2);
  unsigned short* Wwct  = (unsigned short*)alloc((size_t)4 * 64 * 64 * 2);
  unsigned short* Wct   = (unsigned short*)alloc((size_t)4 * 512 * 256 * 2);
  float* stats = (float*)alloc((size_t)28 * 512 * 4);   // sum/sumsq, 28 GEMM slots
  float* scsh  = (float*)alloc((size_t)28 * 512 * 4);   // scale/shift slots
  unsigned short* xbf  = (unsigned short*)alloc((size_t)BDIM * 512 * 2); // input bf16
  unsigned short* abf  = (unsigned short*)alloc((size_t)BDIM * 256 * 2); // l1 / gate-a
  unsigned short* gbbf = (unsigned short*)alloc((size_t)BDIM * 64 * 2);  // gate-b
  unsigned short* h1bf = (unsigned short*)alloc((size_t)BDIM * 256 * 2);
  unsigned short* h2bf = (unsigned short*)alloc((size_t)BDIM * 256 * 2);
  float* pre  = (float*)alloc((size_t)BDIM * 256 * 4);                   // pre-BN f32
  float* ybuf = (float*)alloc((size_t)BDIM * 2 * 4);
  unsigned short* xcat = xbf;  // alias: xbf dead after trunk, reused as [B][512] concat

  hipMemsetAsync(stats, 0, (size_t)28 * 512 * 4, stream);  // zero atomic accumulators

  // ---- weight transpose + bf16 convert ----
  auto trans = [&](const float* s, unsigned short* d, int K, int N) {
    int tot = K * N;
    wtrans<<<(tot + 255) / 256, 256, 0, stream>>>(s, d, K, N);
  };
  trans(W1, W1t, 512, 256);
  trans(W2, W2t, 256, 256);
  for (int k = 0; k < 4; ++k) {
    trans(Wbx1 + (size_t)k * 65536, Wbx1t + (size_t)k * 65536, 256, 256);
    trans(Wbx2 + (size_t)k * 65536, Wbx2t + (size_t)k * 65536, 256, 256);
    trans(Wwa  + (size_t)k * 65536, Wwat  + (size_t)k * 65536, 512, 128);
    trans(Wwb  + (size_t)k * 8192,  Wwbt  + (size_t)k * 8192,  128, 64);
    trans(Wwc  + (size_t)k * 4096,  Wwct  + (size_t)k * 4096,  64, 64);
    trans(Wc   + (size_t)k * 131072, Wct  + (size_t)k * 131072, 512, 256);
  }

  // ---- layer helpers ----
  int slot = 0;
  auto gemmBN = [&](const unsigned short* A, const unsigned short* Wt,
                    const float* bias, float* out, int K, int N,
                    const float* g_, const float* be_) -> float* {
    float* sum  = stats + (size_t)slot * 512;
    float* sum2 = sum + 256;
    dim3 gd(BDIM / 64, N / 64);
    gemm_bias_relu_stats<<<gd, 256, 0, stream>>>(A, Wt, bias, out, sum, sum2, K, N);
    float* sc = scsh + (size_t)slot * 512;
    bn_finalize<<<(N + 63) / 64, 64, 0, stream>>>(sum, sum2, g_, be_, sc, sc + 256,
                                                  N, 1.0f / BDIM);
    ++slot;
    return sc;
  };
  auto apply = [&](const float* src, const float* sc, unsigned short* dst,
                   int N, int dstStride, int dstOff) {
    int tot = BDIM * N;
    bn_apply_bf16<<<(tot + 255) / 256, 256, 0, stream>>>(src, sc, sc + 256, dst,
                                                         N, dstStride, dstOff, tot);
  };

  // ---- shared trunk, both streams ----
  float* sc;
  cvt_f32_bf16<<<(BDIM * 512 + 255) / 256, 256, 0, stream>>>(x1f, xbf, BDIM * 512);
  sc = gemmBN(xbf, W1t, b1, pre, 512, 256, g1, be1);
  apply(pre, sc, abf, 256, 256, 0);
  sc = gemmBN(abf, W2t, b2, pre, 256, 256, g2, be2);
  apply(pre, sc, h1bf, 256, 256, 0);

  cvt_f32_bf16<<<(BDIM * 512 + 255) / 256, 256, 0, stream>>>(x2f, xbf, BDIM * 512);
  sc = gemmBN(xbf, W1t, b1, pre, 512, 256, g1, be1);
  apply(pre, sc, abf, 256, 256, 0);
  sc = gemmBN(abf, W2t, b2, pre, 256, 256, g2, be2);
  apply(pre, sc, h2bf, 256, 256, 0);

  // ---- per-attribute pipeline (sequential over k; buffers reused) ----
  for (int k = 0; k < 4; ++k) {
    // branch projections -> xcat halves
    sc = gemmBN(h1bf, Wbx1t + (size_t)k * 65536, bbx1 + k * 256, pre, 256, 256, g1, be1);
    apply(pre, sc, xcat, 256, 512, 0);
    sc = gemmBN(h2bf, Wbx2t + (size_t)k * 65536, bbx2 + k * 256, pre, 256, 256, g1, be1);
    apply(pre, sc, xcat, 256, 512, 256);

    // gating MLP
    sc = gemmBN(xcat, Wwat + (size_t)k * 65536, bwa + k * 128, pre, 512, 128, g3, be3);
    apply(pre, sc, abf, 128, 128, 0);
    sc = gemmBN(abf, Wwbt + (size_t)k * 8192, bwb + k * 64, pre, 128, 64, g4, be4);
    apply(pre, sc, gbbf, 64, 64, 0);
    sc = gemmBN(gbbf, Wwct + (size_t)k * 4096, bwc + k * 64, pre, 64, 64, g4, be4);
    head_softmax<<<BDIM / 256, 256, 0, stream>>>(pre, sc, sc + 256,
                                                 Wy + (size_t)k * 128, by + k * 2,
                                                 ybuf, 64, 2);

    // weighted concat (in place on xcat), combine layer, output head
    xw_mul<<<(BDIM * 512 + 255) / 256, 256, 0, stream>>>(xcat, ybuf, BDIM * 512);
    sc = gemmBN(xcat, Wct + (size_t)k * 131072, bc + k * 256, pre, 512, 256, g1, be1);
    head_softmax<<<BDIM / 256, 256, 0, stream>>>(pre, sc, sc + 256,
                                                 WoA[k], boA[k],
                                                 (float*)d_out + ooff[k], 256, NoA[k]);
  }
}
